// MoELayer_32822140076136
// MI455X (gfx1250) — matrix operand order repaired and checked by reading, not compiled, not run on hardware
//
#include <hip/hip_runtime.h>
#include <hip/hip_bf16.h>

// ---------------- problem constants ----------------
#define T_TOK   8192      // B*S
#define DIMD    1024
#define NEXP    64
#define HDIM    512
#define CAP     256       // capacity = 1.0 * 2 * T / E
#define KSTEP   64
#define ASTR    72        // LDS row stride (bf16 elems), 144B: 16B aligned, padded
#define BSTR    72

typedef __attribute__((ext_vector_type(16))) __bf16 v16bf;
typedef __attribute__((ext_vector_type(8)))  float  v8f;
typedef __attribute__((ext_vector_type(4)))  int    v4i;

#define AS1 __attribute__((address_space(1)))
#define AS3 __attribute__((address_space(3)))

union Frag16 { uint4 q[2]; v16bf v; };

// ---- CDNA5 async global->LDS staging (ASYNCcnt path), guarded fallback ----
#if defined(__gfx1250__) && __has_builtin(__builtin_amdgcn_global_load_async_to_lds_b128)
#define ASYNC_LDS 1
#else
#define ASYNC_LDS 0
#endif

#if ASYNC_LDS && __has_builtin(__builtin_amdgcn_s_wait_asynccnt)
#define ASYNC_WAIT() __builtin_amdgcn_s_wait_asynccnt(0)
#elif ASYNC_LDS
#define ASYNC_WAIT() asm volatile("s_wait_asynccnt 0x0" ::: "memory")
#else
#define ASYNC_WAIT()
#endif

__device__ __forceinline__ void copy16_g2l(const unsigned short* g, unsigned short* l) {
#if ASYNC_LDS
    __builtin_amdgcn_global_load_async_to_lds_b128(
        (AS1 v4i*)g, (AS3 v4i*)l, 0, 0);
#else
    *(uint4*)l = *(const uint4*)g;
#endif
}

__device__ __forceinline__ unsigned short f2bf(float f) {
    unsigned u = __float_as_uint(f);
    u = (u + 0x7FFFu + ((u >> 16) & 1u)) >> 16;   // round-to-nearest-even
    return (unsigned short)u;
}

// ---------------- 1. router: logits, softmax, top-2 ----------------
__global__ __launch_bounds__(256) void k_gate(const float* __restrict__ x,
                                              const float* __restrict__ wg,
                                              int* __restrict__ topi,
                                              float* __restrict__ gates,
                                              float* __restrict__ probs) {
    __shared__ __align__(16) float swg[128][NEXP];          // 32 KB chunk of wg
    const int tid  = threadIdx.x;
    const int lane = tid & 31;
    const int wid  = tid >> 5;
    const int t0   = blockIdx.x * 16 + wid * 2;             // 2 tokens per wave

    float a00 = 0.f, a01 = 0.f, a10 = 0.f, a11 = 0.f;       // [tok][expert-half]

    for (int c = 0; c < 8; ++c) {
        const int dbase = c * 128;
        // cooperative load of wg[dbase:dbase+128][0:64]
        for (int it = 0; it < 8; ++it) {
            int dl = (tid >> 4) + it * 16;
            int e4 = (tid & 15) * 4;
            float4 v = *(const float4*)&wg[(size_t)(dbase + dl) * NEXP + e4];
            swg[dl][e4 + 0] = v.x; swg[dl][e4 + 1] = v.y;
            swg[dl][e4 + 2] = v.z; swg[dl][e4 + 3] = v.w;
        }
        __syncthreads();
        float xc0[4], xc1[4];
        #pragma unroll
        for (int i = 0; i < 4; ++i) {
            xc0[i] = x[(size_t)t0       * DIMD + dbase + lane + 32 * i];
            xc1[i] = x[(size_t)(t0 + 1) * DIMD + dbase + lane + 32 * i];
        }
        for (int i = 0; i < 4; ++i) {
            #pragma unroll
            for (int s = 0; s < 32; ++s) {
                int dd = i * 32 + s;
                float w0 = swg[dd][lane];
                float w1 = swg[dd][lane + 32];
                float xv0 = __shfl(xc0[i], s, 32);
                float xv1 = __shfl(xc1[i], s, 32);
                a00 = fmaf(xv0, w0, a00); a01 = fmaf(xv0, w1, a01);
                a10 = fmaf(xv1, w0, a10); a11 = fmaf(xv1, w1, a11);
            }
        }
        __syncthreads();
    }

    for (int tk = 0; tk < 2; ++tk) {
        const int t = t0 + tk;
        float l0 = tk ? a10 : a00;
        float l1 = tk ? a11 : a01;
        float m = fmaxf(l0, l1);
        for (int off = 16; off; off >>= 1) m = fmaxf(m, __shfl_xor(m, off, 32));
        float e0 = __expf(l0 - m), e1 = __expf(l1 - m);
        float s = e0 + e1;
        for (int off = 16; off; off >>= 1) s += __shfl_xor(s, off, 32);
        float pr0 = e0 / s, pr1 = e1 / s;
        probs[(size_t)t * NEXP + lane]      = pr0;
        probs[(size_t)t * NEXP + lane + 32] = pr1;

        float v1, v2; int i1, i2;
        if (pr0 >= pr1) { v1 = pr0; i1 = lane;      v2 = pr1; i2 = lane + 32; }
        else            { v1 = pr1; i1 = lane + 32; v2 = pr0; i2 = lane; }
        for (int off = 16; off; off >>= 1) {
            float w1 = __shfl_xor(v1, off, 32); int j1 = __shfl_xor(i1, off, 32);
            float w2 = __shfl_xor(v2, off, 32); int j2 = __shfl_xor(i2, off, 32);
            bool fb = (v1 > w1) || (v1 == w1 && i1 < j1);
            float n1, n2; int k1, k2;
            if (fb) {
                n1 = v1; k1 = i1;
                bool sb = (v2 > w1) || (v2 == w1 && i2 < j1);
                n2 = sb ? v2 : w1; k2 = sb ? i2 : j1;
            } else {
                n1 = w1; k1 = j1;
                bool sb = (w2 > v1) || (w2 == v1 && j2 < i1);
                n2 = sb ? w2 : v1; k2 = sb ? j2 : i1;
            }
            v1 = n1; i1 = k1; v2 = n2; i2 = k2;
        }
        if (lane == 0) {
            topi[t * 2 + 0] = i1; topi[t * 2 + 1] = i2;
            float inv = 1.0f / (v1 + v2);
            gates[t * 2 + 0] = v1 * inv; gates[t * 2 + 1] = v2 * inv;
        }
    }
}

// ---------------- 2. capacity positions (reference ordering) ----------------
__global__ __launch_bounds__(256) void k_pos(const int* __restrict__ topi,
                                             int* __restrict__ pos) {
    const int e    = blockIdx.x;
    const int tid  = threadIdx.x;
    const int lane = tid & 31;
    const int wid  = tid >> 5;
    __shared__ int swt[8];
    __shared__ int srun;
    if (tid == 0) srun = 0;
    __syncthreads();
    for (int k = 0; k < 2; ++k) {               // all k=0 slots precede k=1 slots
        for (int base = 0; base < T_TOK; base += 256) {
            const int t = base + tid;
            const bool match = (topi[t * 2 + k] == e);
            const unsigned mask = (unsigned)__ballot(match);
            const int below = __popc(mask & ((1u << lane) - 1u));
            if (lane == 0) swt[wid] = __popc(mask);
            __syncthreads();
            int wpre = 0, btot = 0;
            for (int w = 0; w < 8; ++w) { if (w < wid) wpre += swt[w]; btot += swt[w]; }
            const int p = srun + wpre + below;
            if (match) pos[t * 2 + k] = p;
            __syncthreads();
            if (tid == 0) srun += btot;
            __syncthreads();
        }
    }
}

// ---------------- 3. dispatch (fp32 -> bf16 scatter) ----------------
__global__ __launch_bounds__(128) void k_dispatch(const float* __restrict__ x,
                                                  const int* __restrict__ topi,
                                                  const int* __restrict__ pos,
                                                  unsigned short* __restrict__ bufx) {
    const int slot = blockIdx.x;                // t*2 + k
    const int p = pos[slot];
    if (p >= CAP) return;                       // dropped token
    const int e = topi[slot];
    const int t = slot >> 1;
    const float* src = x + (size_t)t * DIMD;
    unsigned short* dst = bufx + ((size_t)e * CAP + p) * DIMD;
    for (int i = threadIdx.x; i < DIMD; i += 128) dst[i] = f2bf(src[i]);
}

// ---------------- 4. fused gate+up GEMM + SiLU ----------------
__global__ __launch_bounds__(256) void k_expert_gu(const unsigned short* __restrict__ bufx,
                                                   const float* __restrict__ wgat,
                                                   const float* __restrict__ wup,
                                                   unsigned short* __restrict__ hid) {
    const int e  = blockIdx.y;
    const int mb = (blockIdx.x & 1) * 128;      // C tile
    const int nb = (blockIdx.x >> 1) * 128;     // H tile
    const unsigned short* A = bufx + (size_t)e * CAP * DIMD;
    const float* Bg = wgat + (size_t)e * DIMD * HDIM;
    const float* Bu = wup  + (size_t)e * DIMD * HDIM;

    __shared__ __align__(16) unsigned short sA[128 * ASTR];
    __shared__ __align__(16) unsigned short sBg[128 * BSTR];
    __shared__ __align__(16) unsigned short sBu[128 * BSTR];

    const int tid  = threadIdx.x;
    const int lane = tid & 31;
    const int wid  = tid >> 5;
    const int wm   = (wid & 3) * 32;            // 32 rows per wave
    const int wn   = (wid >> 2) * 64;           // 64 cols per wave
    const int lh   = lane >> 4;                 // lane half selects K sub-range
    const int ln   = lane & 15;

    v8f zero = {};
    v8f accG[2][4], accU[2][4];
    for (int i = 0; i < 2; ++i)
        for (int j = 0; j < 4; ++j) { accG[i][j] = zero; accU[i][j] = zero; }

    for (int kb = 0; kb < DIMD; kb += KSTEP) {
        if (kb + KSTEP < DIMD) {                // stream-in hints for next K panel
            __builtin_prefetch(&Bg[(size_t)(kb + KSTEP) * HDIM + nb], 0, 1);
            __builtin_prefetch(&Bu[(size_t)(kb + KSTEP) * HDIM + nb], 0, 1);
        }
        {   // A tile: 128 x 64 bf16, async global->LDS (ASYNCcnt path)
            int row = tid >> 3;
            const int c8 = (tid & 7) * 8;
            for (int it = 0; it < 4; ++it, row += 32)
                copy16_g2l(&A[(size_t)(mb + row) * DIMD + kb + c8],
                           &sA[row * ASTR + c8]);
        }
        {   // B tiles: fp32 -> bf16, transposed to [n][k] (VGPR path: conversion)
            const int h = tid & 127;
            int kr = tid >> 7;
            for (int it = 0; it < 32; ++it, kr += 2) {
                sBg[h * BSTR + kr] = f2bf(Bg[(size_t)(kb + kr) * HDIM + nb + h]);
                sBu[h * BSTR + kr] = f2bf(Bu[(size_t)(kb + kr) * HDIM + nb + h]);
            }
        }
        ASYNC_WAIT();
        __syncthreads();
        #pragma unroll
        for (int j = 0; j < 2; ++j) {
            const int kk = j * 32;
            Frag16 a[2];
            #pragma unroll
            for (int mt = 0; mt < 2; ++mt) {
                const unsigned short* ap = &sA[(wm + mt * 16 + ln) * ASTR + kk + lh * 16];
                a[mt].q[0] = *(const uint4*)ap;         // K 0..7  (or 16..23)
                a[mt].q[1] = *(const uint4*)(ap + 8);   // K 8..15 (or 24..31)
            }
            #pragma unroll
            for (int nt = 0; nt < 4; ++nt) {
                Frag16 b;
                const unsigned short* bp = &sBg[(wn + nt * 16 + ln) * BSTR + kk + lh * 16];
                b.q[0] = *(const uint4*)bp; b.q[1] = *(const uint4*)(bp + 8);
                #pragma unroll
                for (int mt = 0; mt < 2; ++mt)
                    accG[mt][nt] = __builtin_amdgcn_wmma_f32_16x16x32_bf16(
                        false, a[mt].v, false, b.v, (short)0, accG[mt][nt], false, false);
                const unsigned short* up = &sBu[(wn + nt * 16 + ln) * BSTR + kk + lh * 16];
                b.q[0] = *(const uint4*)up; b.q[1] = *(const uint4*)(up + 8);
                #pragma unroll
                for (int mt = 0; mt < 2; ++mt)
                    accU[mt][nt] = __builtin_amdgcn_wmma_f32_16x16x32_bf16(
                        false, a[mt].v, false, b.v, (short)0, accU[mt][nt], false, false);
            }
        }
        __syncthreads();
    }
    // epilogue: hidden = silu(G) * U  -> bf16
    unsigned short* H = hid + (size_t)e * CAP * HDIM;
    for (int mt = 0; mt < 2; ++mt)
        for (int nt = 0; nt < 4; ++nt) {
            const int n = nb + wn + nt * 16 + ln;
            #pragma unroll
            for (int r = 0; r < 8; ++r) {
                const int m = mb + wm + mt * 16 + r + lh * 8;
                float g = accG[mt][nt][r];
                float u = accU[mt][nt][r];
                float hv = (g / (1.0f + __expf(-g))) * u;
                H[(size_t)m * HDIM + n] = f2bf(hv);
            }
        }
}

// ---------------- 5. down-projection GEMM ----------------
__global__ __launch_bounds__(256) void k_expert_down(const unsigned short* __restrict__ hid,
                                                     const float* __restrict__ wdn,
                                                     float* __restrict__ y) {
    const int e  = blockIdx.y;
    const int mb = (blockIdx.x & 1) * 128;      // C tile
    const int nb = (blockIdx.x >> 1) * 128;     // D tile
    const unsigned short* A = hid + (size_t)e * CAP * HDIM;
    const float* B = wdn + (size_t)e * HDIM * DIMD;

    __shared__ __align__(16) unsigned short sA[128 * ASTR];
    __shared__ __align__(16) unsigned short sB[128 * BSTR];

    const int tid  = threadIdx.x;
    const int lane = tid & 31;
    const int wid  = tid >> 5;
    const int wm   = (wid & 3) * 32;
    const int wn   = (wid >> 2) * 64;
    const int lh   = lane >> 4;
    const int ln   = lane & 15;

    v8f zero = {};
    v8f acc[2][4];
    for (int i = 0; i < 2; ++i) for (int j = 0; j < 4; ++j) acc[i][j] = zero;

    for (int kb = 0; kb < HDIM; kb += KSTEP) {
        if (kb + KSTEP < HDIM)
            __builtin_prefetch(&B[(size_t)(kb + KSTEP) * DIMD + nb], 0, 1);
        {   // A tile (bf16): async global->LDS
            int row = tid >> 3;
            const int c8 = (tid & 7) * 8;
            for (int it = 0; it < 4; ++it, row += 32)
                copy16_g2l(&A[(size_t)(mb + row) * HDIM + kb + c8],
                           &sA[row * ASTR + c8]);
        }
        {
            const int d = tid & 127;
            int kr = tid >> 7;
            for (int it = 0; it < 32; ++it, kr += 2)
                sB[d * BSTR + kr] = f2bf(B[(size_t)(kb + kr) * DIMD + nb + d]);
        }
        ASYNC_WAIT();
        __syncthreads();
        #pragma unroll
        for (int j = 0; j < 2; ++j) {
            const int kk = j * 32;
            Frag16 a[2];
            #pragma unroll
            for (int mt = 0; mt < 2; ++mt) {
                const unsigned short* ap = &sA[(wm + mt * 16 + ln) * ASTR + kk + lh * 16];
                a[mt].q[0] = *(const uint4*)ap;
                a[mt].q[1] = *(const uint4*)(ap + 8);
            }
            #pragma unroll
            for (int nt = 0; nt < 4; ++nt) {
                Frag16 b;
                const unsigned short* bp = &sB[(wn + nt * 16 + ln) * BSTR + kk + lh * 16];
                b.q[0] = *(const uint4*)bp; b.q[1] = *(const uint4*)(bp + 8);
                #pragma unroll
                for (int mt = 0; mt < 2; ++mt)
                    acc[mt][nt] = __builtin_amdgcn_wmma_f32_16x16x32_bf16(
                        false, a[mt].v, false, b.v, (short)0, acc[mt][nt], false, false);
            }
        }
        __syncthreads();
    }
    float* Y = y + (size_t)e * CAP * DIMD;
    for (int mt = 0; mt < 2; ++mt)
        for (int nt = 0; nt < 4; ++nt) {
            const int n = nb + wn + nt * 16 + ln;
            #pragma unroll
            for (int r = 0; r < 8; ++r) {
                const int m = mb + wm + mt * 16 + r + lh * 8;
                Y[(size_t)m * DIMD + n] = acc[mt][nt][r];
            }
        }
}

// ---------------- 6. combine ----------------
__global__ __launch_bounds__(256) void k_combine(const float* __restrict__ y,
                                                 const int* __restrict__ topi,
                                                 const int* __restrict__ pos,
                                                 const float* __restrict__ gates,
                                                 float* __restrict__ out) {
    const int t = blockIdx.x;
    const int e0 = topi[t * 2], e1 = topi[t * 2 + 1];
    int p0 = pos[t * 2], p1 = pos[t * 2 + 1];
    const float w0 = (p0 < CAP) ? gates[t * 2]     : 0.0f;
    const float w1 = (p1 < CAP) ? gates[t * 2 + 1] : 0.0f;
    p0 = min(p0, CAP - 1); p1 = min(p1, CAP - 1);
    const float* r0 = y + ((size_t)e0 * CAP + p0) * DIMD;
    const float* r1 = y + ((size_t)e1 * CAP + p1) * DIMD;
    float* o = out + (size_t)t * DIMD;
    for (int i = threadIdx.x; i < DIMD; i += 256)
        o[i] = w0 * r0[i] + w1 * r1[i];
}

// ---------------- 7. l_aux (deterministic fixed-order reductions) ----------------
__global__ __launch_bounds__(256) void k_aux1(const float* __restrict__ probs,
                                              const int* __restrict__ topi,
                                              float* __restrict__ mece) {
    const int e = blockIdx.x;
    __shared__ float sp[256];
    __shared__ int   sc[256];
    float ps = 0.f; int cs = 0;
    for (int t = threadIdx.x; t < T_TOK; t += 256) {
        ps += probs[(size_t)t * NEXP + e];
        cs += (topi[t * 2] == e) ? 1 : 0;
    }
    sp[threadIdx.x] = ps; sc[threadIdx.x] = cs;
    __syncthreads();
    for (int off = 128; off; off >>= 1) {
        if (threadIdx.x < off) {
            sp[threadIdx.x] += sp[threadIdx.x + off];
            sc[threadIdx.x] += sc[threadIdx.x + off];
        }
        __syncthreads();
    }
    if (threadIdx.x == 0)
        mece[e] = (sp[0] / (float)T_TOK) * ((float)sc[0] / (float)T_TOK);
}

__global__ __launch_bounds__(64) void k_aux2(const float* __restrict__ mece,
                                             float* __restrict__ laux) {
    __shared__ float s[64];
    s[threadIdx.x] = mece[threadIdx.x];
    __syncthreads();
    for (int off = 32; off; off >>= 1) {
        if (threadIdx.x < off) s[threadIdx.x] += s[threadIdx.x + off];
        __syncthreads();
    }
    if (threadIdx.x == 0) *laux = (float)NEXP * s[0];
}

// ---------------- launch ----------------
extern "C" void kernel_launch(void* const* d_in, const int* in_sizes, int n_in,
                              void* d_out, int out_size, void* d_ws, size_t ws_size,
                              hipStream_t stream) {
    const float* x    = (const float*)d_in[0];
    const float* wg   = (const float*)d_in[1];
    const float* wgat = (const float*)d_in[2];
    const float* wup  = (const float*)d_in[3];
    const float* wdn  = (const float*)d_in[4];
    float* out = (float*)d_out;

    char* ws = (char*)d_ws;
    size_t o = 0;
    auto carve = [&](size_t bytes) -> char* {
        o = (o + 255) & ~(size_t)255;
        char* p = ws + o;
        o += bytes;
        return p;
    };
    int*            topi  = (int*)           carve((size_t)T_TOK * 2 * 4);
    int*            pos   = (int*)           carve((size_t)T_TOK * 2 * 4);
    float*          gates = (float*)         carve((size_t)T_TOK * 2 * 4);
    float*          probs = (float*)         carve((size_t)T_TOK * NEXP * 4);
    float*          mece  = (float*)         carve((size_t)NEXP * 4);
    unsigned short* bufx  = (unsigned short*)carve((size_t)NEXP * CAP * DIMD * 2);
    unsigned short* hid   = (unsigned short*)carve((size_t)NEXP * CAP * HDIM * 2);
    float*          y     = (float*)         carve((size_t)NEXP * CAP * DIMD * 4);

    k_gate<<<T_TOK / 16, 256, 0, stream>>>(x, wg, topi, gates, probs);
    k_pos<<<NEXP, 256, 0, stream>>>(topi, pos);
    k_dispatch<<<T_TOK * 2, 128, 0, stream>>>(x, topi, pos, bufx);
    k_expert_gu<<<dim3(8, NEXP), 256, 0, stream>>>(bufx, wgat, wup, hid);
    k_expert_down<<<dim3(16, NEXP), 256, 0, stream>>>(hid, wdn, y);
    k_combine<<<T_TOK, 256, 0, stream>>>(y, topi, pos, gates, out);
    k_aux1<<<NEXP, 256, 0, stream>>>(probs, topi, mece);
    k_aux2<<<1, 64, 0, stream>>>(mece, out + (size_t)T_TOK * DIMD);
}
